// AttentionTorch_75144747811316
// MI455X (gfx1250) — compile-verified, matrix operand order checked by reading, not run on hardware
//
#include <hip/hip_runtime.h>
#include <hip/hip_bf16.h>

#define B_  2
#define S_  2048
#define D_  2048
#define H_  16
#define HD_ 128
#define SCALE_ 0.08838834764831845f   // 1/sqrt(128)

typedef __bf16 bf16;
typedef __attribute__((ext_vector_type(16))) __bf16 v16bf;
typedef __attribute__((ext_vector_type(8)))  __bf16 bf16x8;
typedef __attribute__((ext_vector_type(8)))  float  v8f;

__device__ __forceinline__ v8f wmma_bf16(v16bf a, v16bf b, v8f c) {
  // D = A(16x32) * B(32x16) + C(16x16 f32)
  return __builtin_amdgcn_wmma_f32_16x16x32_bf16(false, a, false, b, (short)0, c,
                                                 false, false);
}

// CDNA5 async copy: 16B per lane, global -> LDS, tracked by ASYNCcnt.
// lds = wave-relative LDS byte offset (low 32 bits of generic shared addr,
// per the flat->LDS address mapping).
__device__ __forceinline__ void async_copy_b128(unsigned lds, const bf16* g) {
  asm volatile("global_load_async_to_lds_b128 %0, %1, off"
               :: "v"(lds), "v"((unsigned long long)(uintptr_t)g)
               : "memory");
}
__device__ __forceinline__ void wait_async0() {
  asm volatile("s_wait_asynccnt 0" ::: "memory");
}
__device__ __forceinline__ unsigned lds_addr(const bf16* p) {
  return (unsigned)(uintptr_t)p;
}

// Row-major fragment loader for a 16x32 bf16 tile at p with leading dim ld
// (elements). Serves the A operand (rows = M) and the B operand when the
// source tile is stored transposed-row-major (rows = N, cols = K) — which is
// exactly how W (y = x*W^T), the K-cache (scores) and transposed-V are laid
// out. ISA 16-bit A layout: element e -> k = half*8 + (e&7) + (e>=8 ? 16 : 0).
__device__ __forceinline__ v16bf load_frag_rm(const bf16* p, int ld) {
  const int lane = threadIdx.x & 31;
  const int m    = lane & 15;
  const int half = lane >> 4;
  const bf16* q = p + m * ld + half * 8;
  bf16x8 lo = *(const bf16x8*)q;
  bf16x8 hi = *(const bf16x8*)(q + 16);
  v16bf f;
#pragma unroll
  for (int e = 0; e < 8; ++e) { f[e] = lo[e]; f[e + 8] = hi[e]; }
  return f;
}

// ---------------------------------------------------------------- convert
__global__ __launch_bounds__(256)
void cvt_kernel(const float* __restrict__ in, bf16* __restrict__ out, size_t n) {
  size_t i = (size_t)blockIdx.x * blockDim.x + threadIdx.x;
  if (i < n) out[i] = (bf16)in[i];
}

// ---------------------------------------------------------------- GEMM
// Y[M,N] = A[M,K] * W[N,K]^T.  128x128 block tile, 8 waves of 32x64.
// K-step 64, double-buffered LDS filled by async-to-LDS copies.
// scatter==1: write bf16 to [B,H,S,HD] head-major; scatter==0: fp32 [M,N].
#define KSTEP 64
#define G_LD  72   // 64 + 8 pad; 144B rows keep 16B alignment
__global__ __launch_bounds__(256)
void gemm_bt_kernel(const bf16* __restrict__ A, const bf16* __restrict__ W,
                    float* __restrict__ outF, bf16* __restrict__ outQ,
                    int M, int N, int K, int scatter) {
  __shared__ bf16 sA[2][128 * G_LD];
  __shared__ bf16 sB[2][128 * G_LD];

  const int tileN = blockIdx.x * 128;
  const int tileM = blockIdx.y * 128;
  const int t    = threadIdx.x;
  const int wave = t >> 5;
  const int lane = t & 31;
  const int wm   = wave & 3;   // 4 m-slots * 32 rows
  const int wn   = wave >> 2;  // 2 n-slots * 64 cols
  const int nl   = lane & 15;
  const int half = lane >> 4;

  v8f acc[2][4];
  const v8f vzero = {0.f,0.f,0.f,0.f,0.f,0.f,0.f,0.f};
#pragma unroll
  for (int i = 0; i < 2; ++i)
#pragma unroll
    for (int j = 0; j < 4; ++j) acc[i][j] = vzero;

  const int r = t >> 1;          // 0..127
  const int c = (t & 1) * 32;    // 0 or 32 elements (64B)

  // Issue async copies of one 128x64 K-slab of A and W into buffer `buf`.
  auto stage = [&](int kk, int buf) {
    const bf16* ga = A + (size_t)(tileM + r) * K + kk * KSTEP + c;
    const bf16* gb = W + (size_t)(tileN + r) * K + kk * KSTEP + c;
    const unsigned la = lds_addr(&sA[buf][r * G_LD + c]);
    const unsigned lb = lds_addr(&sB[buf][r * G_LD + c]);
#pragma unroll
    for (int u = 0; u < 4; ++u) {
      async_copy_b128(la + u * 16, ga + u * 8);
      async_copy_b128(lb + u * 16, gb + u * 8);
    }
    // pull the slab after next into L2 (speculative, safe)
    __builtin_prefetch(ga + KSTEP, 0, 1);
    __builtin_prefetch(gb + KSTEP, 0, 1);
  };

  const int nk = K / KSTEP;
  stage(0, 0);
  wait_async0();
  __syncthreads();

  for (int kk = 0; kk < nk; ++kk) {
    const int cur = kk & 1;
    if (kk + 1 < nk) stage(kk + 1, cur ^ 1);   // overlap copy with WMMAs

#pragma unroll
    for (int kc = 0; kc < 2; ++kc) {
      v16bf af[2], bfv[4];
#pragma unroll
      for (int i = 0; i < 2; ++i)
        af[i] = load_frag_rm(&sA[cur][(wm * 32 + i * 16) * G_LD + kc * 32], G_LD);
#pragma unroll
      for (int j = 0; j < 4; ++j)
        bfv[j] = load_frag_rm(&sB[cur][(wn * 64 + j * 16) * G_LD + kc * 32], G_LD);
#pragma unroll
      for (int i = 0; i < 2; ++i)
#pragma unroll
        for (int j = 0; j < 4; ++j)
          acc[i][j] = wmma_bf16(af[i], bfv[j], acc[i][j]);
    }

    wait_async0();      // next buffer landed
    __syncthreads();    // everyone done reading current buffer
  }

  // Epilogue. C layout: lane -> n = nl (both halves), VGPR rr -> m = rr + 8*half.
#pragma unroll
  for (int i = 0; i < 2; ++i)
#pragma unroll
    for (int j = 0; j < 4; ++j)
#pragma unroll
      for (int rr = 0; rr < 8; ++rr) {
        const int m = tileM + wm * 32 + i * 16 + rr + 8 * half;
        const int n = tileN + wn * 64 + j * 16 + nl;
        const float v = acc[i][j][rr];
        if (scatter) {
          const int b = m / S_, s = m % S_;
          const int h = n / HD_, hd = n % HD_;
          outQ[(((size_t)b * H_ + h) * S_ + s) * HD_ + hd] = (bf16)v;
        } else {
          outF[(size_t)m * N + n] = v;
        }
      }
}

// ---------------------------------------------------------------- attention
// One block = 64 queries of one (b,h); 4 waves, 16 queries each.
// Flash-style online softmax over k-tiles of 64 keys inside the sliding window.
#define A_LDK 136   // 64x(128+8) K tile, rows 272B (16B aligned)
#define A_LDV 72    // 128x(64+8) transposed-V tile, rows 144B
#define A_LDP 72    // per-wave 16x(64+8) P tile
__global__ __launch_bounds__(128)
void attn_kernel(const bf16* __restrict__ Q, const bf16* __restrict__ Kv,
                 const bf16* __restrict__ Vv, const int* __restrict__ amask,
                 const int* __restrict__ winp, bf16* __restrict__ AO) {
  __shared__ bf16 sK [64  * A_LDK];
  __shared__ bf16 sVt[HD_ * A_LDV];
  __shared__ bf16 sP [4 * 16 * A_LDP];
  __shared__ int  sPad[64];

  const int qt   = blockIdx.x;
  const int bh   = blockIdx.y;
  const int b    = bh / H_;
  const int h    = bh % H_;
  const int t    = threadIdx.x;
  const int wave = t >> 5;
  const int lane = t & 31;
  const int nl   = lane & 15;
  const int half = lane >> 4;
  const int win  = winp[0];

  const bf16* Qh = Q  + (size_t)bh * S_ * HD_;
  const bf16* Kh = Kv + (size_t)bh * S_ * HD_;
  const bf16* Vh = Vv + (size_t)bh * S_ * HD_;

  const int q0 = qt * 64 + wave * 16;

  v16bf qf[4];
#pragma unroll
  for (int kc = 0; kc < 4; ++kc)
    qf[kc] = load_frag_rm(Qh + (size_t)q0 * HD_ + kc * 32, HD_);

  const v8f vzero = {0.f,0.f,0.f,0.f,0.f,0.f,0.f,0.f};
  v8f o[8];
  float mrun[8], lrun[8];
#pragma unroll
  for (int ob = 0; ob < 8; ++ob) o[ob] = vzero;
#pragma unroll
  for (int rr = 0; rr < 8; ++rr) { mrun[rr] = -1e30f; lrun[rr] = 0.f; }

  int jmin = qt * 64 - win + 1;
  if (jmin < 0) jmin = 0;
  const int kt0 = jmin >> 6;

  for (int kt = kt0; kt <= qt; ++kt) {
    // ---- stage K tile (async row-major) and V tile (manual transpose)
    {
      const int rr  = t >> 1;         // key row 0..63
      const int off = (t & 1) * 64;   // hd half
      const bf16* gk = Kh + (size_t)(kt * 64 + rr) * HD_ + off;
      const bf16* gv = Vh + (size_t)(kt * 64 + rr) * HD_ + off;
      const unsigned lk = lds_addr(&sK[rr * A_LDK + off]);
#pragma unroll
      for (int u = 0; u < 8; ++u) {
        async_copy_b128(lk + u * 16, gk + u * 8);
        bf16x8 v8 = *(const bf16x8*)(gv + u * 8);
#pragma unroll
        for (int e = 0; e < 8; ++e)
          sVt[(off + u * 8 + e) * A_LDV + rr] = v8[e];
      }
      if (t < 64) sPad[t] = amask[(size_t)b * S_ + kt * 64 + t];
    }
    wait_async0();
    __syncthreads();

    // ---- scores: S(16x64) = Q(16x128) * K^T, 16 WMMAs
    v8f sc[4];
#pragma unroll
    for (int nb = 0; nb < 4; ++nb) sc[nb] = vzero;
#pragma unroll
    for (int kc = 0; kc < 4; ++kc)
#pragma unroll
      for (int nb = 0; nb < 4; ++nb) {
        v16bf kf = load_frag_rm(&sK[(nb * 16) * A_LDK + kc * 32], A_LDK);
        sc[nb] = wmma_bf16(qf[kc], kf, sc[nb]);
      }

    // ---- mask (causal + sliding window + pad) and scale
#pragma unroll
    for (int nb = 0; nb < 4; ++nb) {
      const int jj  = kt * 64 + nb * 16 + nl;
      const int pad = sPad[nb * 16 + nl];
#pragma unroll
      for (int rr = 0; rr < 8; ++rr) {
        const int ii = q0 + rr + 8 * half;
        const bool ok = (jj <= ii) && (jj >= ii - win + 1) && (pad != 0);
        sc[nb][rr] = ok ? sc[nb][rr] * SCALE_ : -__builtin_inff();
      }
    }

    // ---- online softmax; width-16 shuffles reduce each row independently
    float alpha[8];
#pragma unroll
    for (int rr = 0; rr < 8; ++rr) {
      float mx = fmaxf(fmaxf(sc[0][rr], sc[1][rr]), fmaxf(sc[2][rr], sc[3][rr]));
#pragma unroll
      for (int off = 8; off > 0; off >>= 1)
        mx = fmaxf(mx, __shfl_xor(mx, off, 16));
      const float mnew = fmaxf(mrun[rr], mx);
      const float al = __expf(mrun[rr] - mnew);
      mrun[rr] = mnew;
      float rs = 0.f;
#pragma unroll
      for (int nb = 0; nb < 4; ++nb) {
        const float p = __expf(sc[nb][rr] - mnew);
        sc[nb][rr] = p;
        rs += p;
      }
#pragma unroll
      for (int off = 8; off > 0; off >>= 1)
        rs += __shfl_xor(rs, off, 16);
      lrun[rr] = lrun[rr] * al + rs;
      alpha[rr] = al;
    }
#pragma unroll
    for (int ob = 0; ob < 8; ++ob)
#pragma unroll
      for (int rr = 0; rr < 8; ++rr)
        o[ob][rr] *= alpha[rr];

    // ---- C-layout -> A-fragment layout via per-wave LDS P tile
    bf16* Pw = &sP[wave * 16 * A_LDP];
#pragma unroll
    for (int nb = 0; nb < 4; ++nb)
#pragma unroll
      for (int rr = 0; rr < 8; ++rr)
        Pw[(rr + 8 * half) * A_LDP + nb * 16 + nl] = (bf16)sc[nb][rr];

    // ---- O(16x128) += P(16x64) * V(64x128), 16 WMMAs
#pragma unroll
    for (int kc = 0; kc < 2; ++kc) {
      v16bf pf = load_frag_rm(Pw + kc * 32, A_LDP);
#pragma unroll
      for (int ob = 0; ob < 8; ++ob) {
        v16bf vf = load_frag_rm(&sVt[(ob * 16) * A_LDV + kc * 32], A_LDV);
        o[ob] = wmma_bf16(pf, vf, o[ob]);
      }
    }
    __syncthreads();
  }

  // ---- normalize and write attn output in [B,S,D] for the final GEMM
#pragma unroll
  for (int rr = 0; rr < 8; ++rr) {
    const float inv = 1.0f / lrun[rr];
    const int ii = q0 + rr + 8 * half;
#pragma unroll
    for (int ob = 0; ob < 8; ++ob) {
      const int hd = ob * 16 + nl;
      AO[((size_t)b * S_ + ii) * (size_t)D_ + h * HD_ + hd] =
          (bf16)(o[ob][rr] * inv);
    }
  }
}

// ---------------------------------------------------------------- launch
extern "C" void kernel_launch(void* const* d_in, const int* in_sizes, int n_in,
                              void* d_out, int out_size, void* d_ws, size_t ws_size,
                              hipStream_t stream) {
  (void)in_sizes; (void)n_in; (void)out_size; (void)ws_size;
  const float* x  = (const float*)d_in[0];
  const float* Wq = (const float*)d_in[1];
  const float* Wk = (const float*)d_in[2];
  const float* Wv = (const float*)d_in[3];
  const float* Wo = (const float*)d_in[4];
  const int* amask = (const int*)d_in[5];
  const int* winp  = (const int*)d_in[6];
  // d_in[7] = sink_tokens: per reference, sink mask == causal mask; unused.

  const size_t NX = (size_t)B_ * S_ * D_;   // 8.39M
  const size_t NW = (size_t)D_ * D_;        // 4.19M

  bf16* ws  = (bf16*)d_ws;
  bf16* xb  = ws; ws += NX;
  bf16* wqb = ws; ws += NW;
  bf16* wkb = ws; ws += NW;
  bf16* wvb = ws; ws += NW;
  bf16* wob = ws; ws += NW;
  bf16* qb  = ws; ws += NX;
  bf16* kb  = ws; ws += NX;
  bf16* vb  = ws; ws += NX;
  bf16* aob = ws; ws += NX;

  const int thr = 256;
  cvt_kernel<<<(int)((NX + thr - 1) / thr), thr, 0, stream>>>(x,  xb,  NX);
  cvt_kernel<<<(int)((NW + thr - 1) / thr), thr, 0, stream>>>(Wq, wqb, NW);
  cvt_kernel<<<(int)((NW + thr - 1) / thr), thr, 0, stream>>>(Wk, wkb, NW);
  cvt_kernel<<<(int)((NW + thr - 1) / thr), thr, 0, stream>>>(Wv, wvb, NW);
  cvt_kernel<<<(int)((NW + thr - 1) / thr), thr, 0, stream>>>(Wo, wob, NW);

  dim3 ggrid(D_ / 128, (B_ * S_) / 128);
  gemm_bt_kernel<<<ggrid, 256, 0, stream>>>(xb, wqb, nullptr, qb, B_ * S_, D_, D_, 1);
  gemm_bt_kernel<<<ggrid, 256, 0, stream>>>(xb, wkb, nullptr, kb, B_ * S_, D_, D_, 1);
  gemm_bt_kernel<<<ggrid, 256, 0, stream>>>(xb, wvb, nullptr, vb, B_ * S_, D_, D_, 1);

  dim3 agrid(S_ / 64, B_ * H_);
  attn_kernel<<<agrid, 128, 0, stream>>>(qb, kb, vb, amask, winp, aob);

  gemm_bt_kernel<<<ggrid, 256, 0, stream>>>(aob, wob, (float*)d_out, nullptr,
                                            B_ * S_, D_, D_, 0);
}
